// Block_4750233829647
// MI455X (gfx1250) — matrix operand order repaired and checked by reading, compiled, not run on hardware
//
#include <hip/hip_runtime.h>
#include <hip/hip_bf16.h>

typedef __attribute__((ext_vector_type(16))) _Float16 v16h;
typedef __attribute__((ext_vector_type(8)))  float    v8f;
typedef _Float16 h16;

#define NR 110592  // 48*48*48 voxels
#define CCH 128

// ---------------------------------------------------------------- helpers ---

__device__ __forceinline__ v8f wmma32(v16h a, v16h b, v8f c) {
  // D = A(16x32 f16) * B(32x16 f16) + C(16x16 f32)
  return __builtin_amdgcn_wmma_f32_16x16x32_f16(false, a, false, b, (short)0, c,
                                                false, false);
}

// CDNA5 async copy: global -> LDS, 128 bits per lane, tracked by ASYNCcnt.
__device__ __forceinline__ void async_copy_b128(unsigned lds_off, const void* gptr) {
  asm volatile("global_load_async_to_lds_b128 %0, %1, off"
               :
               : "v"(lds_off), "v"((unsigned long long)(uintptr_t)gptr)
               : "memory");
}
__device__ __forceinline__ void async_wait0() {
  asm volatile("s_wait_asynccnt 0x0" ::: "memory");
}

// A fragment: row-major [16 x 32] f16 tile, leading dim lda.
// lanes 0-15 M=lane; element hh holds
// K = (lane/16)*16 + hh
__device__ __forceinline__ v16h frag_a_f16(const h16* A, int lda) {
  int lane = threadIdx.x & 31;
  int m = lane & 15, g = (lane >> 4) & 1;
  v16h f;
#pragma unroll
  for (int hh = 0; hh < 16; ++hh) {
    int v = hh >> 1, j = hh & 1;
    int k = (g << 4) + hh;
    f[hh] = A[m * lda + k];
  }
  return f;
}

// B fragment: row-major [32 x 16] f16 tile (K x N), leading dim ldb.
__device__ __forceinline__ v16h frag_b_f16(const h16* B, int ldb) {
  int lane = threadIdx.x & 31;
  int n = lane & 15, g = (lane >> 4) & 1;
  v16h f;
#pragma unroll
  for (int hh = 0; hh < 16; ++hh) f[hh] = B[(g * 16 + hh) * ldb + n];
  return f;
}

// B fragment from fp32 weights (convert on the fly)
__device__ __forceinline__ v16h frag_b_f32(const float* B, int ldb) {
  int lane = threadIdx.x & 31;
  int n = lane & 15, g = (lane >> 4) & 1;
  v16h f;
#pragma unroll
  for (int hh = 0; hh < 16; ++hh) f[hh] = (h16)B[(g * 16 + hh) * ldb + n];
  return f;
}

// B fragment where logical B[k][n] = Kt[n*ldk + k]  (i.e. B = Kt^T)
__device__ __forceinline__ v16h frag_bT_f16(const h16* Kt, int ldk) {
  int lane = threadIdx.x & 31;
  int n = lane & 15, g = (lane >> 4) & 1;
  v16h f;
#pragma unroll
  for (int hh = 0; hh < 16; ++hh) f[hh] = Kt[n * ldk + g * 16 + hh];
  return f;
}

// window p (0..215), in-window pixel q (0..511) -> voxel row
__device__ __forceinline__ int win_row(int p, int q) {
  int p1 = p / 36, p2 = (p / 6) % 6, p3 = p % 6;
  int y1 = q >> 6, y2 = (q >> 3) & 7, y3 = q & 7;
  return ((p1 * 8 + y1) * 48 + (p2 * 8 + y2)) * 48 + (p3 * 8 + y3);
}

// ------------------------------------------------------------ k1: pos conv --
__global__ void k_posconv(const float* __restrict__ x, const float* __restrict__ w,
                          const float* __restrict__ bias, float* __restrict__ X) {
  int i = blockIdx.x * 256 + threadIdx.x;
  int c = i / NR;
  int r = i - c * NR;
  int zH = r / 2304;
  int rem = r - zH * 2304;
  int zW = rem / 48;
  int zD = rem - zW * 48;
  const float* xc = x + (size_t)c * NR;
  const float* wc = w + c * 27;
  float acc = bias[c];
  for (int dz = 0; dz < 3; ++dz) {
    int az = zH + dz - 1;
    if (az < 0 || az > 47) continue;
    for (int dy = 0; dy < 3; ++dy) {
      int ay = zW + dy - 1;
      if (ay < 0 || ay > 47) continue;
      for (int dx = 0; dx < 3; ++dx) {
        int ax = zD + dx - 1;
        if (ax < 0 || ax > 47) continue;
        acc += xc[(az * 48 + ay) * 48 + ax] * wc[(dz * 3 + dy) * 3 + dx];
      }
    }
  }
  X[(size_t)r * CCH + c] = xc[(zH * 48 + zW) * 48 + zD] + acc;
}

// ------------------------------------------------------------- k2: layernorm
__global__ void k_layernorm(const float* __restrict__ X, const float* __restrict__ w,
                            const float* __restrict__ b, h16* __restrict__ out) {
  int row = blockIdx.x * 8 + (threadIdx.x >> 5);
  int lane = threadIdx.x & 31;
  const float* xr = X + (size_t)row * CCH;
  float4 v = ((const float4*)xr)[lane];
  float s = v.x + v.y + v.z + v.w;
  float ss = v.x * v.x + v.y * v.y + v.z * v.z + v.w * v.w;
#pragma unroll
  for (int m = 16; m >= 1; m >>= 1) {
    s += __shfl_xor(s, m, 32);
    ss += __shfl_xor(ss, m, 32);
  }
  float mu = s * (1.f / 128.f);
  float var = ss * (1.f / 128.f) - mu * mu;
  float rs = rsqrtf(var + 1e-6f);
  int c = lane * 4;
  out[(size_t)row * CCH + c + 0] = (h16)((v.x - mu) * rs * w[c + 0] + b[c + 0]);
  out[(size_t)row * CCH + c + 1] = (h16)((v.y - mu) * rs * w[c + 1] + b[c + 1]);
  out[(size_t)row * CCH + c + 2] = (h16)((v.z - mu) * rs * w[c + 2] + b[c + 2]);
  out[(size_t)row * CCH + c + 3] = (h16)((v.w - mu) * rs * w[c + 3] + b[c + 3]);
}

// ------------------------------------------------------------- k3: qkv GEMM
// C[110592 x 384] = A[110592 x 128](f16) @ W[128 x 384](f32) + bias
__global__ void k_qkv_gemm(const h16* __restrict__ A, const float* __restrict__ W,
                           const float* __restrict__ bias, h16* __restrict__ Qo,
                           h16* __restrict__ Ko, h16* __restrict__ Vo) {
  int row0 = blockIdx.x * 64, col0 = blockIdx.y * 64;
  int wave = threadIdx.x >> 5, lane = threadIdx.x & 31;
#pragma unroll
  for (int tt = 0; tt < 2; ++tt) {
    int t = wave + tt * 8;  // 16 tiles of 16x16
    int tr = t >> 2, tc = t & 3;
    int rbase = row0 + tr * 16, cbase = col0 + tc * 16;
    v8f acc = {};
#pragma unroll
    for (int kk = 0; kk < 4; ++kk) {
      __builtin_prefetch(W + (size_t)(kk * 32) * 384 + cbase, 0, 3);
      v16h a = frag_a_f16(A + (size_t)rbase * CCH + kk * 32, CCH);
      v16h bfr = frag_b_f32(W + (size_t)(kk * 32) * 384 + cbase, 384);
      acc = wmma32(a, bfr, acc);
    }
    int n = lane & 15, hf = (lane >> 4) & 1;
#pragma unroll
    for (int rr = 0; rr < 8; ++rr) {
      int gr = rbase + rr + 8 * hf;
      int gc = cbase + n;
      h16 hv = (h16)(acc[rr] + bias[gc]);
      if (gc < 128)
        Qo[(size_t)gr * CCH + gc] = hv;
      else if (gc < 256)
        Ko[(size_t)gr * CCH + (gc - 128)] = hv;
      else
        Vo[(size_t)gr * CCH + (gc - 256)] = hv;
    }
  }
}

// -------------------------------------------------------- k4: window means --
__global__ void k_win_means(const h16* __restrict__ Q, const h16* __restrict__ K,
                            float* __restrict__ qwin, float* __restrict__ kwin) {
  int p = blockIdx.x, c = threadIdx.x;  // 216 x 128
  float sq = 0.f, sk = 0.f;
  for (int y = 0; y < 512; ++y) {
    int r = win_row(p, y);
    sq += (float)Q[(size_t)r * CCH + c];
    sk += (float)K[(size_t)r * CCH + c];
  }
  qwin[p * CCH + c] = sq * (1.f / 512.f);
  kwin[p * CCH + c] = sk * (1.f / 512.f);
}

// ------------------------------------------------------- k5: routing + topk
__global__ void k_routing_topk(const float* __restrict__ qwin,
                               const float* __restrict__ kwin, int* __restrict__ topi) {
  __shared__ float lg[216];
  int p = blockIdx.x;
  const float scale = 0.0883883476483184f;  // 128^-0.5
  for (int j = threadIdx.x; j < 216; j += blockDim.x) {
    float s = 0.f;
    for (int c = 0; c < CCH; ++c) s += qwin[p * CCH + c] * kwin[j * CCH + c];
    lg[j] = s * scale;
  }
  __syncthreads();
  if (threadIdx.x == 0) {
    int sel[4];
    for (int s = 0; s < 4; ++s) {
      int best = 0;
      float bv = -1e30f;
      for (int j = 0; j < 216; ++j) {
        bool taken = false;
        for (int u = 0; u < s; ++u)
          if (sel[u] == j) taken = true;
        if (!taken && lg[j] > bv) { bv = lg[j]; best = j; }
      }
      sel[s] = best;
      topi[p * 4 + s] = best;
    }
  }
}

// ----------------------------------------------------------- k6: kv pooling
__global__ void k_pool_kv(const h16* __restrict__ K, const h16* __restrict__ V,
                          h16* __restrict__ KP, h16* __restrict__ VP) {
  int blk = blockIdx.x, c = threadIdx.x;  // 216*64 blocks x 128
  int p = blk >> 6, t = blk & 63;
  int a = t >> 4, bb = (t >> 2) & 3, cc = t & 3;
  float sk = 0.f, sv = 0.f;
#pragma unroll
  for (int u = 0; u < 8; ++u) {
    int y1 = 2 * a + (u >> 2), y2 = 2 * bb + ((u >> 1) & 1), y3 = 2 * cc + (u & 1);
    int r = win_row(p, (y1 * 8 + y2) * 8 + y3);
    sk += (float)K[(size_t)r * CCH + c];
    sv += (float)V[(size_t)r * CCH + c];
  }
  KP[(size_t)blk * CCH + c] = (h16)(sk * 0.125f);
  VP[(size_t)blk * CCH + c] = (h16)(sv * 0.125f);
}

// ------------------------------------------------------------- k7: lepe conv
__global__ void k_lepe(const h16* __restrict__ V, const float* __restrict__ w,
                       const float* __restrict__ bias, float* __restrict__ L) {
  int i = blockIdx.x * 256 + threadIdx.x;
  int r = i >> 7, c = i & 127;
  int zH = r / 2304;
  int rem = r - zH * 2304;
  int zW = rem / 48, zD = rem - (rem / 48) * 48;
  const float* wc = w + c * 125;
  float acc = bias[c];
  for (int dz = 0; dz < 5; ++dz) {
    int az = zH + dz - 2;
    if (az < 0 || az > 47) continue;
    for (int dy = 0; dy < 5; ++dy) {
      int ay = zW + dy - 2;
      if (ay < 0 || ay > 47) continue;
      for (int dx = 0; dx < 5; ++dx) {
        int ax = zD + dx - 2;
        if (ax < 0 || ax > 47) continue;
        acc += (float)V[(size_t)((az * 48 + ay) * 48 + ax) * CCH + c] *
               wc[(dz * 5 + dy) * 5 + dx];
      }
    }
  }
  L[(size_t)r * CCH + c] = acc;
}

// ------------------------------------------------------------ k8: attention
// one block = (window p, head m, 64-row q tile); LDS-resident K/V/Q/S/P.
// K/V gather uses CDNA5 async global->LDS copies (ASYNCcnt).
__global__ void k_attention(const h16* __restrict__ Q, const h16* __restrict__ KP,
                            const h16* __restrict__ VP, const int* __restrict__ topi,
                            h16* __restrict__ O) {
  extern __shared__ char smem[];
  h16* sK = (h16*)smem;            // 256 x 32
  h16* sV = sK + 256 * 32;         // 256 x 32
  h16* sQ = sV + 256 * 32;         // 64 x 32
  float* sS = (float*)(sQ + 64 * 32);  // 64 x 256
  h16* sP = (h16*)(sS + 64 * 256);     // 64 x 256
  int qt = blockIdx.x, m = blockIdx.y, p = blockIdx.z;
  int tid = threadIdx.x, lane = tid & 31, wave = tid >> 5;
  int q0 = qt * 64;

  // stage selected pooled K/V via async-to-LDS: 64B (one head) per token,
  // as 4x b128 transfers; 256 tokens * 4 chunks = 1024 per matrix.
  unsigned sKoff = (unsigned)(uintptr_t)sK;  // LDS aperture: addr[31:0]
  unsigned sVoff = (unsigned)(uintptr_t)sV;
  for (int idx = tid; idx < 1024; idx += 256) {
    int kk = idx >> 2, ch = idx & 3;  // token, 16B chunk
    int sel = kk >> 6, t = kk & 63;
    int j = topi[p * 4 + sel];
    size_t src = (size_t)(j * 64 + t) * CCH + m * 32 + ch * 8;
    unsigned dst = (unsigned)(idx * 16);
    async_copy_b128(sKoff + dst, KP + src);
    async_copy_b128(sVoff + dst, VP + src);
  }
  // stage pre-scaled Q (needs f32 scale, so VGPR path)
  const float scale = 0.0883883476483184f;
  for (int idx = tid; idx < 64 * 32; idx += 256) {
    int rr = idx >> 5, cc = idx & 31;
    int r = win_row(p, q0 + rr);
    sQ[idx] = (h16)((float)Q[(size_t)r * CCH + m * 32 + cc] * scale);
  }
  async_wait0();   // drain this wave's ASYNCcnt before the barrier
  __syncthreads();

  // S = Q @ K^T  (64 x 256), 64 WMMA tiles
#pragma unroll
  for (int tt = 0; tt < 8; ++tt) {
    int t = wave + tt * 8;
    int tr = t >> 4, tc = t & 15;
    v16h a = frag_a_f16(sQ + tr * 16 * 32, 32);
    v16h b = frag_bT_f16(sK + tc * 16 * 32, 32);
    v8f acc = {};
    acc = wmma32(a, b, acc);
    int n = lane & 15, hf = (lane >> 4) & 1;
#pragma unroll
    for (int rr = 0; rr < 8; ++rr)
      sS[(tr * 16 + rr + 8 * hf) * 256 + tc * 16 + n] = acc[rr];
  }
  __syncthreads();
  // softmax rows (256 keys)
  for (int row = tid; row < 64; row += 256) {
    float* sr = sS + row * 256;
    float mx = -1e30f;
    for (int j = 0; j < 256; ++j) mx = fmaxf(mx, sr[j]);
    float sum = 0.f;
    for (int j = 0; j < 256; ++j) { float e = __expf(sr[j] - mx); sr[j] = e; sum += e; }
    float inv = 1.f / sum;
    for (int j = 0; j < 256; ++j) sP[row * 256 + j] = (h16)(sr[j] * inv);
  }
  __syncthreads();
  // O = P @ V  (64 x 32), 8 tiles, K-loop of 8
  {
    int tr = wave >> 1, tc = wave & 1;
    v8f acc = {};
#pragma unroll
    for (int kk = 0; kk < 8; ++kk) {
      v16h a = frag_a_f16(sP + tr * 16 * 256 + kk * 32, 256);
      v16h b = frag_b_f16(sV + kk * 32 * 32 + tc * 16, 32);
      acc = wmma32(a, b, acc);
    }
    int n = lane & 15, hf = (lane >> 4) & 1;
#pragma unroll
    for (int rr = 0; rr < 8; ++rr) {
      int r = win_row(p, q0 + tr * 16 + rr + 8 * hf);
      O[(size_t)r * CCH + m * 32 + tc * 16 + n] = (h16)acc[rr];
    }
  }
}

// ------------------------------------------- k9: (attn+lepe)@wo + b + resid
__global__ void k_wo_add(const h16* __restrict__ AO, const float* __restrict__ LEPE,
                         const float* __restrict__ W, const float* __restrict__ bias,
                         float* __restrict__ X) {
  int row0 = blockIdx.x * 64, col0 = blockIdx.y * 64;
  int wave = threadIdx.x >> 5, lane = threadIdx.x & 31;
#pragma unroll
  for (int tt = 0; tt < 2; ++tt) {
    int t = wave + tt * 8;
    int tr = t >> 2, tc = t & 3;
    int rbase = row0 + tr * 16, cbase = col0 + tc * 16;
    v8f acc = {};
#pragma unroll
    for (int kk = 0; kk < 4; ++kk) {
      v16h a;
      {
        int mm = lane & 15, g = (lane >> 4) & 1;
#pragma unroll
        for (int hh = 0; hh < 16; ++hh) {
          int v = hh >> 1, j = hh & 1;
          int k = (g << 4) + hh + kk * 32;
          size_t idx = (size_t)(rbase + mm) * CCH + k;
          a[hh] = (h16)((float)AO[idx] + LEPE[idx]);
        }
      }
      __builtin_prefetch(W + (size_t)(kk * 32) * CCH + cbase, 0, 3);
      v16h b = frag_b_f32(W + (size_t)(kk * 32) * CCH + cbase, CCH);
      acc = wmma32(a, b, acc);
    }
    int n = lane & 15, hf = (lane >> 4) & 1;
#pragma unroll
    for (int rr = 0; rr < 8; ++rr) {
      int gr = rbase + rr + 8 * hf, gc = cbase + n;
      size_t o = (size_t)gr * CCH + gc;
      X[o] = X[o] + acc[rr] + bias[gc];
    }
  }
}

// ------------------------------------- k10: fused MLP + residual + transpose
__global__ void k_mlp(const h16* __restrict__ LN, const float* __restrict__ W1,
                      const float* __restrict__ b1, const float* __restrict__ W2,
                      const float* __restrict__ b2, const float* __restrict__ X,
                      float* __restrict__ out) {
  extern __shared__ char smem[];
  h16* sW = (h16*)smem;            // 128*512 (slot reused for W2 512*128)
  h16* sH = sW + 128 * 512;        // 64*512
  h16* sA = sH + 64 * 512;         // 64*128
  int row0 = blockIdx.x * 64;
  int tid = threadIdx.x, lane = tid & 31, wave = tid >> 5;

  // A tile is contiguous f16: async-copy straight into LDS (16KB = 1024x b128),
  // overlapped with the f32->f16 weight conversion staging below.
  unsigned sAoff = (unsigned)(uintptr_t)sA;
  const h16* gA = LN + (size_t)row0 * CCH;
  for (int idx = tid; idx < 1024; idx += 256)
    async_copy_b128(sAoff + (unsigned)(idx * 16), gA + idx * 8);
  for (int idx = tid; idx < 128 * 512; idx += 256) sW[idx] = (h16)W1[idx];
  async_wait0();
  __syncthreads();

  // H = gelu(A @ W1 + b1)   (64 x 512): 128 WMMA tiles
  for (int tt = 0; tt < 16; ++tt) {
    int t = wave + tt * 8;
    int tr = t >> 5, tc = t & 31;
    v8f acc = {};
#pragma unroll
    for (int kk = 0; kk < 4; ++kk) {
      v16h a = frag_a_f16(sA + tr * 16 * 128 + kk * 32, 128);
      v16h b = frag_b_f16(sW + kk * 32 * 512 + tc * 16, 512);
      acc = wmma32(a, b, acc);
    }
    int n = lane & 15, hf = (lane >> 4) & 1;
#pragma unroll
    for (int rr = 0; rr < 8; ++rr) {
      int gr = tr * 16 + rr + 8 * hf, gc = tc * 16 + n;
      float v2 = acc[rr] + b1[gc];
      float g2 = 0.5f * v2 * (1.f + erff(v2 * 0.7071067811865475f));  // exact gelu
      sH[gr * 512 + gc] = (h16)g2;
    }
  }
  __syncthreads();
  for (int idx = tid; idx < 512 * 128; idx += 256) sW[idx] = (h16)W2[idx];
  __syncthreads();
  // Y = H @ W2 + b2 + X, store NCDHW  (64 x 128): 32 WMMA tiles
  for (int tt = 0; tt < 4; ++tt) {
    int t = wave + tt * 8;
    int tr = t >> 3, tc = t & 7;
    v8f acc = {};
#pragma unroll
    for (int kk = 0; kk < 16; ++kk) {
      v16h a = frag_a_f16(sH + tr * 16 * 512 + kk * 32, 512);
      v16h b = frag_b_f16(sW + kk * 32 * 128 + tc * 16, 128);
      acc = wmma32(a, b, acc);
    }
    int n = lane & 15, hf = (lane >> 4) & 1;
#pragma unroll
    for (int rr = 0; rr < 8; ++rr) {
      int gr = row0 + tr * 16 + rr + 8 * hf, gc = tc * 16 + n;
      float v2 = X[(size_t)gr * CCH + gc] + acc[rr] + b2[gc];
      out[(size_t)gc * NR + gr] = v2;  // NHWDC -> NCDHW transpose on store
    }
  }
}

// ------------------------------------------------------------------- launch
extern "C" void kernel_launch(void* const* d_in, const int* in_sizes, int n_in,
                              void* d_out, int out_size, void* d_ws, size_t ws_size,
                              hipStream_t stream) {
  const float* x = (const float*)d_in[0];
  const float* pos_w = (const float*)d_in[1];
  const float* pos_b = (const float*)d_in[2];
  const float* ln1_w = (const float*)d_in[3];
  const float* ln1_b = (const float*)d_in[4];
  const float* qkv_w = (const float*)d_in[5];
  const float* qkv_b = (const float*)d_in[6];
  const float* lepe_w = (const float*)d_in[7];
  const float* lepe_b = (const float*)d_in[8];
  const float* wo_w = (const float*)d_in[9];
  const float* wo_b = (const float*)d_in[10];
  const float* ln2_w = (const float*)d_in[11];
  const float* ln2_b = (const float*)d_in[12];
  const float* w1 = (const float*)d_in[13];
  const float* b1 = (const float*)d_in[14];
  const float* w2 = (const float*)d_in[15];
  const float* b2 = (const float*)d_in[16];
  float* out = (float*)d_out;

  char* ws = (char*)d_ws;
  const size_t SZ_X = (size_t)NR * CCH * 4;   // 56.6 MB f32
  const size_t SZ_H = (size_t)NR * CCH * 2;   // 28.3 MB f16
  const size_t SZ_P = (size_t)216 * 64 * CCH * 2;
  float* X = (float*)ws;                         // NHWDC fp32 (residual stream)
  h16* LN = (h16*)(ws + SZ_X);                   // ln1/ln2 output f16
  h16* Qh = (h16*)(ws + SZ_X + SZ_H);
  h16* Kh = (h16*)(ws + SZ_X + 2 * SZ_H);        // later reused as attn-out
  h16* Vh = (h16*)(ws + SZ_X + 3 * SZ_H);
  float* LEPE = (float*)(ws + SZ_X + 4 * SZ_H);  // 56.6 MB f32
  h16* KP = (h16*)(ws + 2 * SZ_X + 4 * SZ_H);
  h16* VP = (h16*)(ws + 2 * SZ_X + 4 * SZ_H + SZ_P);
  float* QW = (float*)(ws + 2 * SZ_X + 4 * SZ_H + 2 * SZ_P);
  float* KW = QW + 216 * CCH;
  int* TOPI = (int*)(KW + 216 * CCH);

  k_posconv<<<55296, 256, 0, stream>>>(x, pos_w, pos_b, X);
  k_layernorm<<<13824, 256, 0, stream>>>(X, ln1_w, ln1_b, LN);
  k_qkv_gemm<<<dim3(1728, 6), 256, 0, stream>>>(LN, qkv_w, qkv_b, Qh, Kh, Vh);
  k_win_means<<<216, 128, 0, stream>>>(Qh, Kh, QW, KW);
  k_routing_topk<<<216, 256, 0, stream>>>(QW, KW, TOPI);
  k_pool_kv<<<13824, 128, 0, stream>>>(Kh, Vh, KP, VP);
  k_lepe<<<55296, 256, 0, stream>>>(Vh, lepe_w, lepe_b, LEPE);

  const int attn_lds = 135168;  // sK 16K + sV 16K + sQ 4K + sS 64K + sP 32K
  (void)hipFuncSetAttribute((const void*)k_attention,
                            hipFuncAttributeMaxDynamicSharedMemorySize, attn_lds);
  k_attention<<<dim3(8, 4, 216), 256, attn_lds, stream>>>(Qh, KP, VP, TOPI, Kh);

  k_wo_add<<<dim3(1728, 2), 256, 0, stream>>>(Kh, LEPE, wo_w, wo_b, X);
  k_layernorm<<<13824, 256, 0, stream>>>(X, ln2_w, ln2_b, LN);

  const int mlp_lds = 212992;  // sW 128K + sH 64K + sA 16K
  (void)hipFuncSetAttribute((const void*)k_mlp,
                            hipFuncAttributeMaxDynamicSharedMemorySize, mlp_lds);
  k_mlp<<<1728, 256, mlp_lds, stream>>>(LN, w1, b1, w2, b2, X, out);
}
